// QwenAttention_9345848836830
// MI455X (gfx1250) — matrix operand order repaired and checked by reading, not compiled, not run on hardware
//
#include <hip/hip_runtime.h>
#include <stdint.h>

#define B_  2
#define S_  2048
#define H_  2048
#define NH_ 16
#define HD_ 128
#define K_  2048
#define M_  4096   // B_*S_

typedef __attribute__((ext_vector_type(16))) __bf16 v16bf;
typedef __attribute__((ext_vector_type(8)))  float  v8f;
typedef __attribute__((ext_vector_type(4)))  unsigned int u32x4;
typedef __attribute__((ext_vector_type(8)))  int i32x8;
typedef __attribute__((ext_vector_type(4)))  int i32x4;

union AFrag { v16bf v; uint4 q[2]; };
union BFrag { v16bf v; uint4 q[2]; unsigned u[8]; };

__device__ __forceinline__ unsigned short f2bf_u(float f) {
  unsigned u = __builtin_bit_cast(unsigned, f);
  unsigned r = u + 0x7fffu + ((u >> 16) & 1u);   // round-to-nearest-even
  return (unsigned short)(r >> 16);
}
__device__ __forceinline__ __bf16 f2bf(float f) {
  unsigned short h = f2bf_u(f);
  return __builtin_bit_cast(__bf16, h);
}
__device__ __forceinline__ unsigned pk2(float lo, float hi) {
  return ((unsigned)f2bf_u(hi) << 16) | (unsigned)f2bf_u(lo);
}
__device__ __forceinline__ v8f wmma_bf16(const v16bf& a, const v16bf& b, const v8f& c) {
  return __builtin_amdgcn_wmma_f32_16x16x32_bf16(false, a, false, b, (short)0, c, false, false);
}
__device__ __forceinline__ unsigned lds_off(const void* p) {
  // generic LDS pointer: addr[31:0] is the LDS byte offset (ISA flat aperture rules)
  return (unsigned)(unsigned long long)(uintptr_t)p;
}

// --------------------------------------------------------- TDM 2-D tile load
// Loads tile_d1 rows x tile_d0 elements (2 bytes each, contiguous per row,
// row pitch = row_stride elements) from global into LDS at lds_addr, inserting
// pad DWORDs after each (2<<pi) DWORDs per ISA D# group1 encoding:
//   pad_interval code pi: pad after 2<<pi DWORDs; pad_amount code pa: (pa+1) DWORDs.
__device__ __forceinline__ void tdm_load_2d(unsigned lds_addr, const void* gptr,
                                            unsigned tile_d0, unsigned tile_d1,
                                            unsigned long long row_stride,
                                            unsigned pi, unsigned pa) {
  unsigned long long ga = (unsigned long long)(uintptr_t)gptr;
  u32x4 g0;
  g0[0] = 1u;                                             // count=1, no gather
  g0[1] = lds_addr;                                       // lds_addr
  g0[2] = (unsigned)(ga & 0xffffffffu);                   // global_addr lo
  g0[3] = (unsigned)((ga >> 32) & 0x01ffffffu) | (2u << 30);  // ga hi | type=2
  i32x8 g1;
  g1[0] = (int)((1u << 16) | (1u << 20) | (pi << 22) | (pa << 25)); // ds=2B, pad_en
  g1[1] = (int)((tile_d0 & 0xffffu) << 16);               // tensor_dim0[15:0]
  g1[2] = (int)(((tile_d0 >> 16) & 0xffffu) | ((tile_d1 & 0xffffu) << 16));
  g1[3] = (int)(((tile_d1 >> 16) & 0xffffu) | ((tile_d0 & 0xffffu) << 16)); // tile_dim0
  g1[4] = (int)(tile_d1 & 0xffffu);                       // tile_dim1, tile_dim2=0
  g1[5] = (int)(unsigned)(row_stride & 0xffffffffu);      // dim0_stride lo
  g1[6] = (int)(unsigned)((row_stride >> 32) & 0xffffu);  // dim0_stride hi
  g1[7] = 0;
  i32x4 z4 = {0, 0, 0, 0};
  i32x8 z8 = {0, 0, 0, 0, 0, 0, 0, 0};
  __builtin_amdgcn_tensor_load_to_lds(g0, g1, z4, z4, z8, 0);
}

// ---------------------------------------------------------------- fp32 -> bf16
__global__ __launch_bounds__(256) void cvt_bf16(const float* __restrict__ x,
                                                __bf16* __restrict__ y, int n) {
  int i = (blockIdx.x * 256 + threadIdx.x) * 4;
  if (i < n) {
    float4 f = *(const float4*)(x + i);
    uint2 o;
    o.x = pk2(f.x, f.y);
    o.y = pk2(f.z, f.w);
    *(uint2*)(y + i) = o;
  }
}

// -------------------------------------------- W[k][n] fp32 -> WT[n][k] bf16
__global__ __launch_bounds__(256) void transpose_w(const float* __restrict__ W,
                                                   __bf16* __restrict__ WT) {
  __shared__ float tile[32][33];
  const int bx = blockIdx.x * 32;   // n
  const int by = blockIdx.y * 32;   // k
  const int tx = threadIdx.x & 31;
  const int ty = threadIdx.x >> 5;  // 0..7
#pragma unroll
  for (int i = 0; i < 32; i += 8)
    tile[ty + i][tx] = W[(size_t)(by + ty + i) * H_ + bx + tx];
  __syncthreads();
#pragma unroll
  for (int i = 0; i < 32; i += 8)
    WT[(size_t)(bx + ty + i) * K_ + by + tx] = f2bf(tile[tx][ty + i]);
}

// ----------------------------------------------------------------- GEMM
// C[M,N] = A[M,K](bf16,row-major) * B (via BT[n][k] bf16) + bias
// MODE 0: store bf16 scattered [b][h][s][d]   (Q, K)
// MODE 2: store bf16 scattered [b][h][d][s]   (V transposed)
// MODE 3: store fp32 row-major [M][N] + bias  (output projection)
#define LDSA 40   // 32 bf16 + 8 pad (80B rows; TDM pad: pi=3 (64B), pa=3 (16B))
template <int MODE>
__global__ __launch_bounds__(256) void gemm128(const __bf16* __restrict__ A,
                                               const __bf16* __restrict__ BT,
                                               const float* __restrict__ bias,
                                               void* __restrict__ out) {
  __shared__ __align__(16) __bf16 sA[2][128 * LDSA];
  __shared__ __align__(16) __bf16 sB[2][128 * LDSA];
  const int tid = threadIdx.x, lane = tid & 31, wid = tid >> 5;
  const int g = lane >> 4, lr = lane & 15;
  const int bm = blockIdx.y * 128, bn = blockIdx.x * 128;
  const int wm = (wid >> 2) * 64, wn = (wid & 3) * 32;
  const int NKT = K_ / 32;

  v8f acc[4][2];
  const v8f zero8 = {0.f, 0.f, 0.f, 0.f, 0.f, 0.f, 0.f, 0.f};
#pragma unroll
  for (int i = 0; i < 4; i++)
#pragma unroll
    for (int j = 0; j < 2; j++) acc[i][j] = zero8;

  // prologue: TDM-load tile 0 into buffer 0 (wave 0 drives the DMA)
  if (wid == 0) {
    tdm_load_2d(lds_off(sA[0]), A + (size_t)bm * K_, 32, 128, K_, 3, 3);
    tdm_load_2d(lds_off(sB[0]), BT + (size_t)bn * K_, 32, 128, K_, 3, 3);
  }

  for (int kt = 0; kt < NKT; ++kt) {
    const int cur = kt & 1;
    if (wid == 0 && kt + 1 < NKT) {  // issue next tile into the other buffer
      const int kk = (kt + 1) * 32;
      tdm_load_2d(lds_off(sA[cur ^ 1]), A + (size_t)bm * K_ + kk, 32, 128, K_, 3, 3);
      tdm_load_2d(lds_off(sB[cur ^ 1]), BT + (size_t)bn * K_ + kk, 32, 128, K_, 3, 3);
    }
    if (kt + 1 < NKT) __builtin_amdgcn_s_wait_tensorcnt(2);  // current tile retired
    else              __builtin_amdgcn_s_wait_tensorcnt(0);
    __syncthreads();

    // load ALL fragments first, then a dense WMMA chain
    AFrag af[4]; BFrag bfr[2];
#pragma unroll
    for (int i = 0; i < 4; i++) {
      const char* p = (const char*)(sA[cur] + (wm + i * 16 + lr) * LDSA) + g * 32;
      af[i].q[0] = *(const uint4*)p;
      af[i].q[1] = *(const uint4*)(p + 16);
    }
#pragma unroll
    for (int j = 0; j < 2; j++) {
      const char* p = (const char*)(sB[cur] + (wn + j * 16 + lr) * LDSA) + g * 32;
      bfr[j].q[0] = *(const uint4*)p;
      bfr[j].q[1] = *(const uint4*)(p + 16);
    }
#pragma unroll
    for (int i = 0; i < 4; i++)
#pragma unroll
      for (int j = 0; j < 2; j++)
        acc[i][j] = wmma_bf16(af[i].v, bfr[j].v, acc[i][j]);
    __syncthreads();  // readers done before this buffer is overwritten
  }

#pragma unroll
  for (int i = 0; i < 4; i++)
#pragma unroll
    for (int j = 0; j < 2; j++) {
      const int n = bn + wn + j * 16 + lr;
      const float bv = bias[n];
#pragma unroll
      for (int r = 0; r < 8; r++) {
        const int m = bm + wm + i * 16 + r + 8 * g;
        const float v = acc[i][j][r] + bv;
        if (MODE == 3) {
          ((float*)out)[(size_t)m * H_ + n] = v;
        } else {
          const int b = m >> 11, s = m & (S_ - 1);
          const int h = n >> 7, d = n & (HD_ - 1);
          if (MODE == 0)
            ((__bf16*)out)[(((size_t)(b * NH_ + h)) * S_ + s) * HD_ + d] = f2bf(v);
          else  // MODE 2: V transposed [b][h][d][s]
            ((__bf16*)out)[(((size_t)(b * NH_ + h)) * HD_ + d) * S_ + s] = f2bf(v);
        }
      }
    }
}

// ------------------------------------------------------------ flash attention
// Per block: one (b,h), 128 q rows. Per wave: 16 q columns (transposed math).
//   S^T = K * Q^T  (A = K tile, B = Q rows)      -> softmax state per lane (q)
//   O^T = V^T * P^T (A = Vt tile, B via shfl)
// K/V tiles streamed by the Tensor Data Mover, double-buffered.
#define LDSK 136  // 128 bf16 + 8 pad (272B rows; TDM pad: pi=5 (256B), pa=3)
__global__ __launch_bounds__(256) void flash_attn(const __bf16* __restrict__ Q,
                                                  const __bf16* __restrict__ Kt,
                                                  const __bf16* __restrict__ Vt,
                                                  __bf16* __restrict__ O) {
  __shared__ __align__(16) __bf16 sK[2][32 * LDSK];   // [32 keys][128 d]
  __shared__ __align__(16) __bf16 sV[2][128 * LDSA];  // [128 d][32 keys]
  const int tid = threadIdx.x, lane = tid & 31, wid = tid >> 5;
  const int g = lane >> 4, lr = lane & 15;
  const int bh = blockIdx.y;              // b*NH + h
  const int qt = blockIdx.x;
  const int q_glob = qt * 128 + wid * 16 + lr;

  const __bf16* Qh = Q + (size_t)bh * S_ * HD_;
  const __bf16* Kh = Kt + (size_t)bh * S_ * HD_;
  const __bf16* Vh = Vt + (size_t)bh * HD_ * S_;

  // loop-invariant Q B-fragments (4 chunks of d)
  BFrag qf[4];
#pragma unroll
  for (int c = 0; c < 4; c++) {
    const char* p = (const char*)(Qh + (size_t)q_glob * HD_) + c * 64 + g * 32;
    qf[c].q[0] = *(const uint4*)p;
    qf[c].q[1] = *(const uint4*)(p + 16);
  }

  const v8f zero8 = {0.f, 0.f, 0.f, 0.f, 0.f, 0.f, 0.f, 0.f};
  v8f oacc[8];
#pragma unroll
  for (int j = 0; j < 8; j++) oacc[j] = zero8;
  float mrow = -3.0e38f, lrow = 0.f;
  const float sc = 0.08838834764831845f * 1.4426950408889634f;  // 1/sqrt(128)*log2(e)

  const int nkt = (qt + 1) * 4;  // 32-key tiles (causal upper bound for block)

  if (wid == 0) {  // prologue: tile 0 -> buffer 0
    tdm_load_2d(lds_off(sK[0]), Kh, 128, 32, HD_, 5, 3);
    tdm_load_2d(lds_off(sV[0]), Vh, 32, 128, S_, 3, 3);
  }

  for (int kt = 0; kt < nkt; ++kt) {
    const int kbase = kt * 32;
    const int cur = kt & 1;
    if (wid == 0 && kt + 1 < nkt) {
      const int kb2 = kbase + 32;
      tdm_load_2d(lds_off(sK[cur ^ 1]), Kh + (size_t)kb2 * HD_, 128, 32, HD_, 5, 3);
      tdm_load_2d(lds_off(sV[cur ^ 1]), Vh + kb2, 32, 128, S_, 3, 3);
    }
    if (kt + 1 < nkt) __builtin_amdgcn_s_wait_tensorcnt(2);
    else              __builtin_amdgcn_s_wait_tensorcnt(0);
    __syncthreads();

    // load all 8 K fragments, then run the 8 score WMMAs back-to-back
    AFrag kf[8];
#pragma unroll
    for (int c = 0; c < 4; c++) {
      const char* p0 = (const char*)(sK[cur] + lr * LDSK) + c * 64 + g * 32;
      kf[c].q[0] = *(const uint4*)p0;
      kf[c].q[1] = *(const uint4*)(p0 + 16);
      const char* p1 = (const char*)(sK[cur] + (16 + lr) * LDSK) + c * 64 + g * 32;
      kf[4 + c].q[0] = *(const uint4*)p1;
      kf[4 + c].q[1] = *(const uint4*)(p1 + 16);
    }
    v8f sa = zero8, sb = zero8;
#pragma unroll
    for (int c = 0; c < 4; c++) sa = wmma_bf16(kf[c].v, qf[c].v, sa);
#pragma unroll
    for (int c = 0; c < 4; c++) sb = wmma_bf16(kf[4 + c].v, qf[c].v, sb);

    // causal mask + online softmax (state per lane == per q column)
    float p[16];
    float tmax = -3.0e38f;
#pragma unroll
    for (int r = 0; r < 8; r++) {
      const int k0 = kbase + r + 8 * g;
      const float v0 = (k0 <= q_glob) ? sa[r] * sc : -3.0e38f;
      const float v1 = (k0 + 16 <= q_glob) ? sb[r] * sc : -3.0e38f;
      p[r] = v0;  p[r + 8] = v1;
      tmax = fmaxf(tmax, fmaxf(v0, v1));
    }
    tmax = fmaxf(tmax, __shfl_xor(tmax, 16));
    const float mnew = fmaxf(mrow, tmax);
    const float alpha = exp2f(mrow - mnew);
    float rsum = 0.f;
#pragma unroll
    for (int e = 0; e < 16; e++) { p[e] = exp2f(p[e] - mnew); rsum += p[e]; }
    rsum += __shfl_xor(rsum, 16);
    lrow = lrow * alpha + rsum;
    mrow = mnew;
#pragma unroll
    for (int j = 0; j < 8; j++)
#pragma unroll
      for (int r = 0; r < 8; r++) oacc[j][r] *= alpha;

    // build P^T B-fragment via shuffles (no LDS round-trip)
    unsigned pkA[4], pkB[4];
#pragma unroll
    for (int i = 0; i < 4; i++) {
      pkA[i] = pk2(p[2 * i], p[2 * i + 1]);
      pkB[i] = pk2(p[8 + 2 * i], p[9 + 2 * i]);
    }
    BFrag pf;
#pragma unroll
    for (int d = 0; d < 8; d++) {
      const int src = ((d >> 2) << 4) + lr;
      const unsigned x0 = (unsigned)__shfl((int)pkA[d & 3], src, 32);
      const unsigned x1 = (unsigned)__shfl((int)pkB[d & 3], src, 32);
      pf.u[d] = g ? x1 : x0;
    }

    // load all 8 V fragments, then the 8 O^T WMMAs back-to-back
    AFrag vf[8];
#pragma unroll
    for (int j = 0; j < 8; j++) {
      const char* p2 = (const char*)(sV[cur] + (j * 16 + lr) * LDSA) + g * 32;
      vf[j].q[0] = *(const uint4*)p2;
      vf[j].q[1] = *(const uint4*)(p2 + 16);
    }
#pragma unroll
    for (int j = 0; j < 8; j++) oacc[j] = wmma_bf16(vf[j].v, pf.v, oacc[j]);
    __syncthreads();  // readers done before this buffer is overwritten
  }

  const float inv = 1.0f / lrow;
  const int b = bh >> 4, h = bh & (NH_ - 1);
  __bf16* Orow = O + (size_t)(b * S_ + q_glob) * H_ + h * HD_;
#pragma unroll
  for (int j = 0; j < 8; j++) {
    uint4 w;
    unsigned* wp = (unsigned*)&w;
#pragma unroll
    for (int r = 0; r < 4; r++)
      wp[r] = pk2(oacc[j][2 * r] * inv, oacc[j][2 * r + 1] * inv);
    *(uint4*)(Orow + j * 16 + 8 * g) = w;  // d = 16j + 8g + 0..7
  }
}

// ---------------------------------------------------------------- launcher
extern "C" void kernel_launch(void* const* d_in, const int* in_sizes, int n_in,
                              void* d_out, int out_size, void* d_ws, size_t ws_size,
                              hipStream_t stream) {
  const float* X  = (const float*)d_in[0];
  // d_in[1] = attention_mask (causal; applied analytically)
  const float* Wq = (const float*)d_in[2]; const float* bq = (const float*)d_in[3];
  const float* Wk = (const float*)d_in[4]; const float* bk = (const float*)d_in[5];
  const float* Wv = (const float*)d_in[6]; const float* bv = (const float*)d_in[7];
  const float* Wo = (const float*)d_in[8]; const float* bo = (const float*)d_in[9];
  float* out = (float*)d_out;

  char* w = (char*)d_ws;
  const size_t MB = 1024 * 1024;
  __bf16* Xb  = (__bf16*)(w + 0);          // 16 MB  [M][K]
  __bf16* WTq = (__bf16*)(w + 16 * MB);    //  8 MB  [N][K]
  __bf16* WTk = (__bf16*)(w + 24 * MB);
  __bf16* WTv = (__bf16*)(w + 32 * MB);
  __bf16* WTo = (__bf16*)(w + 40 * MB);
  __bf16* Qb  = (__bf16*)(w + 48 * MB);    // 16 MB  [b][h][s][d]
  __bf16* Kb  = (__bf16*)(w + 64 * MB);    // 16 MB  [b][h][s][d]
  __bf16* Vtb = (__bf16*)(w + 80 * MB);    // 16 MB  [b][h][d][s]
  __bf16* Ab  = (__bf16*)(w + 96 * MB);    // 16 MB  [M][H]

  // 1) convert hidden states to bf16
  cvt_bf16<<<(M_ * H_) / (256 * 4), 256, 0, stream>>>(X, Xb, M_ * H_);

  // 2) transpose+convert weights
  dim3 tg(H_ / 32, K_ / 32);
  transpose_w<<<tg, 256, 0, stream>>>(Wq, WTq);
  transpose_w<<<tg, 256, 0, stream>>>(Wk, WTk);
  transpose_w<<<tg, 256, 0, stream>>>(Wv, WTv);
  transpose_w<<<tg, 256, 0, stream>>>(Wo, WTo);

  // 3) QKV projections (WMMA bf16, TDM-staged)
  dim3 gg(H_ / 128, M_ / 128);
  gemm128<0><<<gg, 256, 0, stream>>>(Xb, WTq, bq, (void*)Qb);
  gemm128<0><<<gg, 256, 0, stream>>>(Xb, WTk, bk, (void*)Kb);
  gemm128<2><<<gg, 256, 0, stream>>>(Xb, WTv, bv, (void*)Vtb);

  // 4) causal flash attention (TDM-staged K/V)
  dim3 ag(S_ / 128, B_ * NH_);
  flash_attn<<<ag, 256, 0, stream>>>(Qb, Kb, Vtb, Ab);

  // 5) output projection, fp32 + bias
  gemm128<3><<<gg, 256, 0, stream>>>(Ab, WTo, bo, (void*)out);
}
